// BahdanauAttention_30743375905297
// MI455X (gfx1250) — matrix operand order repaired and checked by reading, not compiled, not run on hardware
//
#include <hip/hip_runtime.h>
#include <math.h>

#define B_  32
#define T_  2048
#define D_  1024
#define U_  1024
#define NT_ 4          // t-tiles (of 16 rows) per workgroup, double-buffered

typedef __bf16 bf16_t;
typedef __attribute__((ext_vector_type(16))) __bf16 v16bf;
typedef __attribute__((ext_vector_type(8)))  float  v8f;

// gcc-style int4 to match the async-LDS builtin's parameter type exactly
typedef int v4i_t __attribute__((vector_size(16)));
typedef __attribute__((address_space(1))) v4i_t* gbl_v4i;
typedef __attribute__((address_space(3))) v4i_t* lds_v4i;

#if __has_builtin(__builtin_amdgcn_global_load_async_to_lds_b128)
#define HAVE_ASYNC_LDS 1
#else
#define HAVE_ASYNC_LDS 0
#endif

// Copy one 16x1024 fp32 tile (row-padded to 1028) global -> LDS.
// Async path uses GLOBAL_LOAD_ASYNC_TO_LDS_B128 (ASYNCcnt); fallback is
// synchronous float4 staging.
__device__ __forceinline__ void stage_tile(float* dst_base,
                                           const float* src_base) {
#pragma unroll 4
  for (int it = 0; it < 16; ++it) {
    const int idx = it * 1024 + (int)threadIdx.x * 4;
    const int row = idx >> 10;
    const int col = idx & (D_ - 1);
#if HAVE_ASYNC_LDS
    __builtin_amdgcn_global_load_async_to_lds_b128(
        (gbl_v4i)(src_base + idx),
        (lds_v4i)(dst_base + row * 1028 + col),
        0, 0);
#else
    float4 v = *(const float4*)(src_base + idx);
    *(float4*)(dst_base + row * 1028 + col) = v;
#endif
  }
}

__device__ __forceinline__ void wait_async_prev() {  // previous tile complete
#if HAVE_ASYNC_LDS
#if __has_builtin(__builtin_amdgcn_s_wait_asynccnt)
  __builtin_amdgcn_s_wait_asynccnt(16);
#else
  asm volatile("s_wait_asynccnt 0x10" ::: "memory");
#endif
#endif
}

__device__ __forceinline__ void wait_async_all() {   // everything complete
#if HAVE_ASYNC_LDS
#if __has_builtin(__builtin_amdgcn_s_wait_asynccnt)
  __builtin_amdgcn_s_wait_asynccnt(0);
#else
  asm volatile("s_wait_asynccnt 0x0" ::: "memory");
#endif
#endif
}

// ---------------------------------------------------------------------------
// Kernel 1: q_proj[b][u] = query[b]·W1[:,u] + b1[u] + b2[u]   (tiny GEMM)
// ---------------------------------------------------------------------------
__global__ __launch_bounds__(256) void qproj_kernel(
    const float* __restrict__ query, const float* __restrict__ W1,
    const float* __restrict__ b1, const float* __restrict__ b2,
    float* __restrict__ qp) {
  const int b = blockIdx.y;
  const int u = blockIdx.x * 256 + threadIdx.x;
  const float* q = query + (size_t)b * D_;
  float acc = b1[u] + b2[u];
#pragma unroll 4
  for (int d = 0; d < D_; ++d)
    acc = fmaf(q[d], W1[(size_t)d * U_ + u], acc);
  qp[(size_t)b * U_ + u] = acc;
}

// ---------------------------------------------------------------------------
// Kernel 2: w2t[u][d] = bf16(W2[d][u])  (LDS-tiled transpose + convert)
// ---------------------------------------------------------------------------
__global__ __launch_bounds__(256) void w2t_kernel(
    const float* __restrict__ W2, bf16_t* __restrict__ w2t) {
  __shared__ float tile[32][33];
  const int u0 = blockIdx.x * 32;
  const int d0 = blockIdx.y * 32;
  const int tx = threadIdx.x & 31;
  const int ty = threadIdx.x >> 5;  // 0..7
#pragma unroll
  for (int i = 0; i < 4; ++i)
    tile[ty + i * 8][tx] = W2[(size_t)(d0 + ty + i * 8) * U_ + (u0 + tx)];
  __syncthreads();
#pragma unroll
  for (int i = 0; i < 4; ++i)
    w2t[(size_t)(u0 + ty + i * 8) * D_ + (d0 + tx)] =
        (bf16_t)tile[tx][ty + i * 8];
}

// ---------------------------------------------------------------------------
// Kernel 3 (main): persistent over NT_ t-tiles with double-buffered async
// LDS staging; scores = tanh(values_tile @ W2 + qproj); logit = scores @ V
// via v_wmma_f32_16x16x32_bf16 (8 waves x 8 N-tiles cover U=1024).
// ---------------------------------------------------------------------------
__global__ __launch_bounds__(256) void score_kernel(
    const float* __restrict__ values, const bf16_t* __restrict__ w2t,
    const float* __restrict__ qp, const float* __restrict__ Vp,
    const float* __restrict__ bv, float* __restrict__ logits) {
  // fp32 tiles, row stride 1028 => lane m's ds_load_b128 hits banks 4m..4m+3
  __shared__ __align__(16) float sbuf[2][16 * 1028];
  __shared__ float slog[16];

  const int b     = blockIdx.y;
  const int tbase = blockIdx.x * (NT_ * 16);
  const float* srcb = values + ((size_t)b * T_ + tbase) * D_;

  const int lane = threadIdx.x & 31;
  const int wave = threadIdx.x >> 5;   // owns U columns [wave*128, +128)
  const int h    = lane >> 4;          // half-group
  const int mrow = lane & 15;          // A row; N-within-tile for B/C/D
  const float bv0 = bv[0];

  // B fragment base: lane(h, n) holds K = h*16 + (0..15), contiguous in w2t.
  const bf16_t* wbase = w2t + (size_t)(wave * 128 + mrow) * D_ + h * 16;

  // Per-column qproj and V are tile-invariant: load once.
  float qv[8], vv[8];
#pragma unroll
  for (int nt = 0; nt < 8; ++nt) {
    const int n = wave * 128 + nt * 16 + mrow;
    qv[nt] = qp[(size_t)b * U_ + n];
    vv[nt] = Vp[n];
  }

  stage_tile(sbuf[0], srcb);  // prologue: tile 0 in flight

#pragma unroll 1
  for (int i = 0; i < NT_; ++i) {
    if (i + 1 < NT_) {  // overlap next tile's HBM traffic with this tile's WMMA
      stage_tile(sbuf[(i + 1) & 1], srcb + (size_t)(i + 1) * 16 * D_);
      wait_async_prev();
    } else {
      wait_async_all();
    }
    if (threadIdx.x < 16) slog[threadIdx.x] = 0.0f;
    __syncthreads();

    const float* abase = &sbuf[i & 1][mrow * 1028 + h * 16];
    v8f acc[8] = {};

#pragma unroll 2
    for (int kk = 0; kk < D_; kk += 32) {
      // A fragment: lane(h, m): elems 0..15 <- K=kk+h*16+e
      union { v16bf v; bf16_t e[16]; } A;
      const float* a0 = abase + kk;
      const float* a1 = abase + kk + 8;
#pragma unroll
      for (int e = 0; e < 8; ++e) {
        A.e[e]     = (bf16_t)a0[e];
        A.e[e + 8] = (bf16_t)a1[e];
      }
      if (kk + 32 < D_)  // warm next K-chunk of the (L2-resident) B stream
        __builtin_prefetch((const void*)(wbase + kk + 32), 0, 3);
#pragma unroll
      for (int nt = 0; nt < 8; ++nt) {
        v16bf Bf = *(const v16bf*)(wbase + (size_t)nt * 16 * D_ + kk);
        acc[nt] = __builtin_amdgcn_wmma_f32_16x16x32_bf16(
            false, A.v, false, Bf, (short)0, acc[nt], false, false);
      }
    }

    // Epilogue: s = tanh(acc + qproj[n]); partial_logit[row] += s * V[n]
    // C/D layout: VGPR r, lane -> (row = r + 8*h, col = mrow within tile).
    float p[8] = {};
#pragma unroll
    for (int nt = 0; nt < 8; ++nt) {
#pragma unroll
      for (int r = 0; r < 8; ++r) {
        float s = tanhf(acc[nt][r] + qv[nt]);
        p[r] = fmaf(s, vv[nt], p[r]);
      }
    }
#pragma unroll
    for (int off = 1; off < 16; off <<= 1) {
#pragma unroll
      for (int r = 0; r < 8; ++r) p[r] += __shfl_xor(p[r], off, 32);
    }
    if (mrow == 0) {
#pragma unroll
      for (int r = 0; r < 8; ++r) atomicAdd(&slog[h * 8 + r], p[r]);
    }
    __syncthreads();
    if (threadIdx.x < 16)
      logits[(size_t)b * T_ + tbase + i * 16 + threadIdx.x] =
          slog[threadIdx.x] + bv0;
    __syncthreads();  // safe to overwrite sbuf[i&1] next iteration
  }
}

// ---------------------------------------------------------------------------
// Kernel 4: softmax over T per batch -> attention weights
// ---------------------------------------------------------------------------
__global__ __launch_bounds__(256) void softmax_kernel(
    const float* __restrict__ logits, float* __restrict__ weights) {
  __shared__ float sred[256];
  const int b = blockIdx.x, tid = threadIdx.x;
  const float* L = logits + (size_t)b * T_;

  float mx = -3.4e38f;
  for (int t = tid; t < T_; t += 256) mx = fmaxf(mx, L[t]);
  sred[tid] = mx;
  __syncthreads();
  for (int s = 128; s > 0; s >>= 1) {
    if (tid < s) sred[tid] = fmaxf(sred[tid], sred[tid + s]);
    __syncthreads();
  }
  mx = sred[0];
  __syncthreads();

  float sum = 0.0f;
  for (int t = tid; t < T_; t += 256) sum += __expf(L[t] - mx);
  sred[tid] = sum;
  __syncthreads();
  for (int s = 128; s > 0; s >>= 1) {
    if (tid < s) sred[tid] += sred[tid + s];
    __syncthreads();
  }
  const float inv = 1.0f / sred[0];
  for (int t = tid; t < T_; t += 256)
    weights[(size_t)b * T_ + t] = __expf(L[t] - mx) * inv;
}

// ---------------------------------------------------------------------------
// Kernel 5: context[b][d] = sum_t w[b][t] * values[b][t][d]  (bandwidth-bound)
// ---------------------------------------------------------------------------
__global__ __launch_bounds__(256) void context_kernel(
    const float* __restrict__ values, const float* __restrict__ weights,
    float* __restrict__ context) {
  __shared__ float sw[T_];
  const int b = blockIdx.y;
  const int d = blockIdx.x * 256 + threadIdx.x;
  for (int t = threadIdx.x; t < T_; t += 256)
    sw[t] = weights[(size_t)b * T_ + t];
  __syncthreads();
  const float* vb = values + (size_t)b * T_ * D_ + d;
  float acc = 0.0f;
#pragma unroll 8
  for (int t = 0; t < T_; ++t)
    acc = fmaf(sw[t], vb[(size_t)t * D_], acc);
  context[(size_t)b * D_ + d] = acc;
}

// ---------------------------------------------------------------------------
extern "C" void kernel_launch(void* const* d_in, const int* in_sizes, int n_in,
                              void* d_out, int out_size, void* d_ws, size_t ws_size,
                              hipStream_t stream) {
  const float* query  = (const float*)d_in[0];
  const float* values = (const float*)d_in[1];
  const float* W1     = (const float*)d_in[2];
  const float* b1     = (const float*)d_in[3];
  const float* W2     = (const float*)d_in[4];
  const float* b2     = (const float*)d_in[5];
  const float* V      = (const float*)d_in[6];
  const float* bv     = (const float*)d_in[7];

  float* out     = (float*)d_out;
  float* context = out;               // [B, D]
  float* weights = out + B_ * D_;     // [B, T, 1]

  char*   ws     = (char*)d_ws;
  float*  qp     = (float*)ws;                                   // 128 KB
  bf16_t* w2t    = (bf16_t*)(ws + 128 * 1024);                   // 2 MB
  float*  logits = (float*)(ws + 128 * 1024 + 2 * 1024 * 1024);  // 256 KB

  qproj_kernel  <<<dim3(U_ / 256, B_), 256, 0, stream>>>(query, W1, b1, b2, qp);
  w2t_kernel    <<<dim3(U_ / 32, D_ / 32), 256, 0, stream>>>(W2, w2t);
  score_kernel  <<<dim3(T_ / (16 * NT_), B_), 256, 0, stream>>>(values, w2t, qp, V, bv, logits);
  softmax_kernel<<<B_, 256, 0, stream>>>(logits, weights);
  context_kernel<<<dim3(D_ / 256, B_), 256, 0, stream>>>(values, weights, context);
}
